// SparseOutputProjection_38431367364876
// MI455X (gfx1250) — matrix operand order repaired and checked by reading, not compiled, not run on hardware
//
#include <hip/hip_runtime.h>
#include <math.h>

#define TOK_TOTAL 4096
#define SEQ 2048
#define DHID 1024
#define VOCAB 16000

typedef __attribute__((ext_vector_type(16))) __bf16 v16bf;
typedef __attribute__((ext_vector_type(8)))  float  v8f;
typedef __attribute__((ext_vector_type(4)))  int    v4i;

union FragBF {
  v16bf v;
  uint4 q[2];
};

// ---------------------------------------------------------------------------
// CDNA5 async global->LDS copy (ASYNCcnt) with graceful fallback
// ---------------------------------------------------------------------------
#if defined(__has_builtin)
#if __has_builtin(__builtin_amdgcn_global_load_async_to_lds_b128)
#define HAS_ASYNC_LDS 1
#endif
#endif
#ifndef HAS_ASYNC_LDS
#define HAS_ASYNC_LDS 0
#endif

#if HAS_ASYNC_LDS
__device__ __forceinline__ void async_copy_b128(const __bf16* g, __bf16* l) {
  __attribute__((address_space(1))) v4i* gp = (__attribute__((address_space(1))) v4i*)g;
  __attribute__((address_space(3))) v4i* lp = (__attribute__((address_space(3))) v4i*)l;
  __builtin_amdgcn_global_load_async_to_lds_b128(gp, lp, 0, 0);
}
#if __has_builtin(__builtin_amdgcn_s_wait_asynccnt)
#define WAIT_ASYNC(n) __builtin_amdgcn_s_wait_asynccnt(n)
#else
#define WAIT_ASYNC(n) asm volatile("s_wait_asynccnt %0" :: "n"(n) : "memory")
#endif
#endif

// ---------------------------------------------------------------------------
// Tier counter reset
// ---------------------------------------------------------------------------
__global__ void zero_counts_kernel(int* counts) {
  if (threadIdx.x < 8) counts[threadIdx.x] = 0;
}

// ---------------------------------------------------------------------------
// Per-token routing: tok_attn = sum(attn), per-batch max, threshold, and
// compaction of token indices into 3 tier lists. One block per batch.
// ---------------------------------------------------------------------------
__global__ void strategy_kernel(const float* __restrict__ attn,
                                int* __restrict__ counts,
                                int* __restrict__ lists) {
  const int b = blockIdx.x;
  const float* ab = attn + (size_t)b * SEQ * 16;
  __shared__ float red[256];

  float lmax = -3.4e38f;
  for (int s = threadIdx.x; s < SEQ; s += 256) {
    float t = 0.f;
#pragma unroll
    for (int j = 0; j < 16; ++j) t += ab[s * 16 + j];
    lmax = fmaxf(lmax, t);
  }
  red[threadIdx.x] = lmax;
  __syncthreads();
  for (int o = 128; o > 0; o >>= 1) {
    if (threadIdx.x < o) red[threadIdx.x] = fmaxf(red[threadIdx.x], red[threadIdx.x + o]);
    __syncthreads();
  }
  const float denom = red[0] + 1e-8f;

  for (int s = threadIdx.x; s < SEQ; s += 256) {
    float t = 0.f;
#pragma unroll
    for (int j = 0; j < 16; ++j) t += ab[s * 16 + j];
    const float na = t / denom;
    const int strat = (na >= 0.7f) ? 2 : ((na >= 0.3f) ? 1 : 0);
    const int pos = atomicAdd(&counts[strat], 1);
    lists[strat * TOK_TOTAL + pos] = b * SEQ + s;
  }
}

// ---------------------------------------------------------------------------
// f32 -> bf16 elementwise convert
// ---------------------------------------------------------------------------
__global__ void convert_bf16_kernel(const float* __restrict__ in,
                                    __bf16* __restrict__ out,
                                    long nValid, long nTotal) {
  long i = ((long)blockIdx.x * blockDim.x + threadIdx.x) * 4;
#pragma unroll
  for (int j = 0; j < 4; ++j) {
    long k = i + j;
    if (k < nTotal) out[k] = (k < nValid) ? (__bf16)in[k] : (__bf16)0.0f;
  }
}

// ---------------------------------------------------------------------------
// f32 [K,N] -> bf16 [N,Kpad] transpose-convert (zero pad k in [K,Kpad)).
// ---------------------------------------------------------------------------
__global__ void transpose_bf16_kernel(const float* __restrict__ W,
                                      __bf16* __restrict__ WT,
                                      int K, int N, int Kpad) {
  __shared__ float tile[32][33];
  const int k0 = blockIdx.x * 32;
  const int n0 = blockIdx.y * 32;
  const int tx = threadIdx.x;
  const int ty = threadIdx.y;
#pragma unroll
  for (int i = 0; i < 4; ++i) {
    const int k = k0 + ty + i * 8;
    const int n = n0 + tx;
    tile[ty + i * 8][tx] = (k < K && n < N) ? W[(size_t)k * N + n] : 0.0f;
  }
  __syncthreads();
#pragma unroll
  for (int i = 0; i < 4; ++i) {
    const int n = n0 + ty + i * 8;
    const int k = k0 + tx;
    if (n < N && k < Kpad) WT[(size_t)n * Kpad + k] = (__bf16)tile[tx][ty + i * 8];
  }
}

// ---------------------------------------------------------------------------
// bf16 WMMA GEMM: C[M,N] = A[M,Kp] * B^T (B stored [N,Kp]) (+bias, epilogue)
// Block: 256 threads / 8 waves. Block tile 64x128, wave tile 32x32, K-step 32.
// MODE 0: hidden linear  -> bf16 store to Hout
// MODE 1: hidden + GELU  -> bf16 store to Hout
// MODE 2: tier logits    -> gathered rows (idxList/cntPtr), f32 store to Cout
// MODE 3: focus          -> Cout[m,n] *= 1 + sigmoid(acc + bias)
// ---------------------------------------------------------------------------
template <int MODE>
__global__ __launch_bounds__(256)
void gemm_bf16_wmma(const __bf16* __restrict__ A,
                    const __bf16* __restrict__ B,
                    const float* __restrict__ bias,
                    int M, int Nlog, int Kp,
                    __bf16* __restrict__ Hout, int ldh, int storeN,
                    float* __restrict__ Cout,
                    const int* __restrict__ idxList,
                    const int* __restrict__ cntPtr) {
  constexpr int BM = 64, BN = 128, KS = 32;
  constexpr int ASTR = 40;              // padded LDS row stride: 80B, bank-stride 20
  constexpr int BSTR = 40;
  constexpr int AELEMS = BM * ASTR;     // elems per A buffer
  constexpr int BELEMS = BN * BSTR;

#if HAS_ASYNC_LDS
  __shared__ __align__(16) __bf16 Alds[2 * AELEMS];   // double-buffered
  __shared__ __align__(16) __bf16 Blds[2 * BELEMS];
#else
  __shared__ __align__(16) __bf16 Alds[AELEMS];
  __shared__ __align__(16) __bf16 Blds[BELEMS];
#endif

  const int m0 = blockIdx.y * BM;
  const int n0 = blockIdx.x * BN;
  int cnt = M;
  if (MODE == 2) {
    cnt = *cntPtr;
    if (m0 >= cnt) return;   // uniform early-exit for over-provisioned tier grids
  }

  const int tid  = threadIdx.x;
  const int lane = tid & 31;
  const int wave = tid >> 5;
  const int lo = lane & 15;
  const int hi = lane >> 4;
  const int wm = wave & 1;    // wave M half (0..1)
  const int wn = wave >> 1;   // wave N quarter (0..3)

  // --- per-thread global->LDS assignments (fixed for whole K loop) ---
  const int rowA = tid >> 2;            // 0..63
  const int chA  = tid & 3;
  int gRow = m0 + rowA;
  if (MODE == 2) gRow = idxList[gRow < cnt ? gRow : (cnt - 1)];
  const __bf16* aSrc = A + (size_t)gRow * Kp + chA * 8;

  const int chB   = tid & 3;
  const int nRow0 = tid >> 2;           // 0..63
  const int nRow1 = 64 + nRow0;         // 64..127
  // clamp instead of zero-fill: garbage columns are never stored
  const int nIdx0 = (n0 + nRow0 < Nlog) ? (n0 + nRow0) : (Nlog - 1);
  const int nIdx1 = (n0 + nRow1 < Nlog) ? (n0 + nRow1) : (Nlog - 1);
  const __bf16* bSrc0 = B + (size_t)nIdx0 * Kp + chB * 8;
  const __bf16* bSrc1 = B + (size_t)nIdx1 * Kp + chB * 8;

  __bf16* aDst  = &Alds[rowA * ASTR + chA * 8];
  __bf16* bDst0 = &Blds[nRow0 * BSTR + chB * 8];
  __bf16* bDst1 = &Blds[nRow1 * BSTR + chB * 8];

  // --- per-wave LDS fragment offsets ---
  int aOff[2], bOff[2];
#pragma unroll
  for (int r = 0; r < 2; ++r) aOff[r] = (wm * 32 + r * 16 + lo) * ASTR + hi * 8;
#pragma unroll
  for (int c = 0; c < 2; ++c) bOff[c] = (wn * 32 + c * 16 + lo) * BSTR + hi * 8;

  v8f acc[2][2] = {};
  const int nIter = Kp / KS;

#if HAS_ASYNC_LDS
  // ------------- async double-buffered pipeline (ASYNCcnt) -------------
  {
    async_copy_b128(aSrc, aDst);
    async_copy_b128(bSrc0, bDst0);
    async_copy_b128(bSrc1, bDst1);
  }
  for (int it = 0; it < nIter; ++it) {
    const int cur = it & 1;
    if (it + 1 < nIter) {
      const int koff = (it + 1) * KS;
      const int nxt = cur ^ 1;
      async_copy_b128(aSrc + koff, aDst + nxt * AELEMS);
      async_copy_b128(bSrc0 + koff, bDst0 + nxt * BELEMS);
      async_copy_b128(bSrc1 + koff, bDst1 + nxt * BELEMS);
      WAIT_ASYNC(3);   // tile `it` (first 3 ops, in order) complete
    } else {
      WAIT_ASYNC(0);
    }
    __syncthreads();   // all waves' writes for tile `it` visible

    const int ab = cur * AELEMS;
    const int bb = cur * BELEMS;
    FragBF a[2], b[2];
#pragma unroll
    for (int r = 0; r < 2; ++r) {
      a[r].q[0] = *(const uint4*)&Alds[ab + aOff[r]];
      a[r].q[1] = *(const uint4*)&Alds[ab + aOff[r] + 16];
    }
#pragma unroll
    for (int c = 0; c < 2; ++c) {
      b[c].q[0] = *(const uint4*)&Blds[bb + bOff[c]];
      b[c].q[1] = *(const uint4*)&Blds[bb + bOff[c] + 16];
    }
#pragma unroll
    for (int r = 0; r < 2; ++r)
#pragma unroll
      for (int c = 0; c < 2; ++c)
        acc[r][c] = __builtin_amdgcn_wmma_f32_16x16x32_bf16(
            false, a[r].v, false, b[c].v, (short)0, acc[r][c], false, false);
    __syncthreads();   // reads of buf `cur` done before it is refilled next iter
  }
#else
  // ------------- register-prefetch software pipeline -------------
  uint4 ra  = *(const uint4*)(aSrc);
  uint4 rb0 = *(const uint4*)(bSrc0);
  uint4 rb1 = *(const uint4*)(bSrc1);
  for (int it = 0; it < nIter; ++it) {
    __syncthreads();
    *(uint4*)aDst  = ra;
    *(uint4*)bDst0 = rb0;
    *(uint4*)bDst1 = rb1;
    __syncthreads();
    if (it + 1 < nIter) {
      const int koff = (it + 1) * KS;
      ra  = *(const uint4*)(aSrc + koff);   // in flight during WMMAs below
      rb0 = *(const uint4*)(bSrc0 + koff);
      rb1 = *(const uint4*)(bSrc1 + koff);
    }
    FragBF a[2], b[2];
#pragma unroll
    for (int r = 0; r < 2; ++r) {
      a[r].q[0] = *(const uint4*)&Alds[aOff[r]];
      a[r].q[1] = *(const uint4*)&Alds[aOff[r] + 16];
    }
#pragma unroll
    for (int c = 0; c < 2; ++c) {
      b[c].q[0] = *(const uint4*)&Blds[bOff[c]];
      b[c].q[1] = *(const uint4*)&Blds[bOff[c] + 16];
    }
#pragma unroll
    for (int r = 0; r < 2; ++r)
#pragma unroll
      for (int c = 0; c < 2; ++c)
        acc[r][c] = __builtin_amdgcn_wmma_f32_16x16x32_bf16(
            false, a[r].v, false, b[c].v, (short)0, acc[r][c], false, false);
  }
#endif

  // ---- epilogue: C layout lane -> N=lo, VGPR g -> M = g + 8*hi ----
#pragma unroll
  for (int r = 0; r < 2; ++r) {
#pragma unroll
    for (int c = 0; c < 2; ++c) {
      const int n = n0 + wn * 32 + c * 16 + lo;
      float bval = 0.0f;
      if (MODE != 2 || bias != nullptr) {
        if (n < Nlog) bval = bias ? bias[n] : 0.0f;
      }
#pragma unroll
      for (int g = 0; g < 8; ++g) {
        const int mloc = m0 + wm * 32 + r * 16 + g + 8 * hi;
        float v = acc[r][c][g] + bval;
        if (MODE <= 1) {
          if (MODE == 1) v = 0.5f * v * (1.0f + erff(v * 0.70710678118654752f));
          if (n < storeN) Hout[(size_t)mloc * ldh + n] = (__bf16)((n < Nlog) ? v : 0.0f);
        } else if (MODE == 2) {
          if (mloc < cnt) {
            const int row = idxList[mloc];
            Cout[(size_t)row * VOCAB + n] = v;
          }
        } else {
          const float f = 1.0f + 1.0f / (1.0f + __expf(-v));
          const size_t o = (size_t)mloc * VOCAB + n;
          Cout[o] = Cout[o] * f;
        }
      }
    }
  }
}

// ---------------------------------------------------------------------------
extern "C" void kernel_launch(void* const* d_in, const int* in_sizes, int n_in,
                              void* d_out, int out_size, void* d_ws, size_t ws_size,
                              hipStream_t stream) {
  (void)in_sizes; (void)n_in; (void)out_size; (void)ws_size;
  const float* hs     = (const float*)d_in[0];
  const float* attn   = (const float*)d_in[1];
  const float* W_full = (const float*)d_in[2];
  const float* W_s1   = (const float*)d_in[3];
  const float* b_s1   = (const float*)d_in[4];
  const float* W_s2   = (const float*)d_in[5];
  const float* b_s2   = (const float*)d_in[6];
  const float* W_m1   = (const float*)d_in[7];
  const float* b_m1   = (const float*)d_in[8];
  const float* W_me   = (const float*)d_in[9];
  const float* b_me   = (const float*)d_in[10];
  const float* W_f1   = (const float*)d_in[11];
  const float* b_f1   = (const float*)d_in[12];
  const float* W_f2   = (const float*)d_in[13];
  const float* b_f2   = (const float*)d_in[14];
  float* out = (float*)d_out;

  char* base = (char*)d_ws;
  size_t off = 0;
  auto carveB = [&](size_t bytes) -> char* {
    char* p = base + off;
    off = (off + bytes + 255) & ~(size_t)255;
    return p;
  };
  int* counts = (int*)carveB(16 * sizeof(int));
  int* lists  = (int*)carveB((size_t)3 * TOK_TOTAL * sizeof(int));
  auto carve = [&](size_t elems) -> __bf16* { return (__bf16*)carveB(elems * 2); };

  __bf16* hB   = carve((size_t)TOK_TOTAL * DHID);
  __bf16* WfT  = carve((size_t)VOCAB * DHID);
  __bf16* Ws1T = carve((size_t)4000 * DHID);
  __bf16* Ws2T = carve((size_t)VOCAB * 4000);
  __bf16* Wm1T = carve((size_t)2000 * DHID);
  __bf16* WmeT = carve((size_t)VOCAB * 2016);
  __bf16* Wf1T = carve((size_t)1600 * DHID);
  __bf16* Wf2T = carve((size_t)VOCAB * 1600);
  __bf16* Hs   = carve((size_t)TOK_TOTAL * 4000);
  __bf16* Hm   = carve((size_t)TOK_TOTAL * 2016);
  __bf16* Hf   = carve((size_t)TOK_TOTAL * 1600);

  // 1) routing
  zero_counts_kernel<<<1, 32, 0, stream>>>(counts);
  strategy_kernel<<<2, 256, 0, stream>>>(attn, counts, lists);

  // 2) precision conversion / layout
  {
    long n = (long)TOK_TOTAL * DHID;
    convert_bf16_kernel<<<(int)((n + 1023) / 1024), 256, 0, stream>>>(hs, hB, n, n);
  }
  dim3 tb(32, 8);
  transpose_bf16_kernel<<<dim3(DHID / 32, VOCAB / 32), tb, 0, stream>>>(W_full, WfT, DHID, VOCAB, DHID);
  transpose_bf16_kernel<<<dim3(DHID / 32, 4000 / 32), tb, 0, stream>>>(W_s1, Ws1T, DHID, 4000, DHID);
  transpose_bf16_kernel<<<dim3(4000 / 32, VOCAB / 32), tb, 0, stream>>>(W_s2, Ws2T, 4000, VOCAB, 4000);
  transpose_bf16_kernel<<<dim3(DHID / 32, (2000 + 31) / 32), tb, 0, stream>>>(W_m1, Wm1T, DHID, 2000, DHID);
  transpose_bf16_kernel<<<dim3(2016 / 32, VOCAB / 32), tb, 0, stream>>>(W_me, WmeT, 2000, VOCAB, 2016);
  transpose_bf16_kernel<<<dim3(DHID / 32, 1600 / 32), tb, 0, stream>>>(W_f1, Wf1T, DHID, 1600, DHID);
  transpose_bf16_kernel<<<dim3(1600 / 32, VOCAB / 32), tb, 0, stream>>>(W_f2, Wf2T, 1600, VOCAB, 1600);

  // 3) hidden projections
  gemm_bf16_wmma<1><<<dim3(32, 64), 256, 0, stream>>>(hB, Ws1T, b_s1, TOK_TOTAL, 4000, DHID,
                                                      Hs, 4000, 4000, nullptr, nullptr, nullptr);
  gemm_bf16_wmma<0><<<dim3(16, 64), 256, 0, stream>>>(hB, Wm1T, b_m1, TOK_TOTAL, 2000, DHID,
                                                      Hm, 2016, 2016, nullptr, nullptr, nullptr);
  gemm_bf16_wmma<1><<<dim3(13, 64), 256, 0, stream>>>(hB, Wf1T, b_f1, TOK_TOTAL, 1600, DHID,
                                                      Hf, 1600, 1600, nullptr, nullptr, nullptr);

  // 4) tier logits (gathered rows; blocks past tier count exit immediately)
  gemm_bf16_wmma<2><<<dim3(125, 64), 256, 0, stream>>>(Hm, WmeT, b_me, TOK_TOTAL, VOCAB, 2016,
                                                       nullptr, 0, 0, out, lists + 0 * TOK_TOTAL, counts + 0);
  gemm_bf16_wmma<2><<<dim3(125, 64), 256, 0, stream>>>(Hs, Ws2T, b_s2, TOK_TOTAL, VOCAB, 4000,
                                                       nullptr, 0, 0, out, lists + 1 * TOK_TOTAL, counts + 1);
  gemm_bf16_wmma<2><<<dim3(125, 64), 256, 0, stream>>>(hB, WfT, nullptr, TOK_TOTAL, VOCAB, DHID,
                                                       nullptr, 0, 0, out, lists + 2 * TOK_TOTAL, counts + 2);

  // 5) vocabulary focusing, fused sigmoid-scale RMW on the logits
  gemm_bf16_wmma<3><<<dim3(125, 64), 256, 0, stream>>>(Hf, Wf2T, b_f2, TOK_TOTAL, VOCAB, 1600,
                                                       nullptr, 0, 0, out, nullptr, nullptr);
}
